// CrossAttention_68006512164917
// MI455X (gfx1250) — matrix operand order repaired and checked by reading, not compiled, not run on hardware
//
#include <hip/hip_runtime.h>

// ---------- types ----------
typedef __bf16 bf16;
typedef bf16  v16bf __attribute__((ext_vector_type(16)));
typedef bf16  v8bf  __attribute__((ext_vector_type(8)));
typedef bf16  v4bf  __attribute__((ext_vector_type(4)));
typedef float v8f   __attribute__((ext_vector_type(8)));
typedef float v4f32 __attribute__((ext_vector_type(4)));
typedef unsigned int  uint4a __attribute__((ext_vector_type(4)));
typedef int           int4a  __attribute__((ext_vector_type(4)));
typedef int           int8a  __attribute__((ext_vector_type(8)));

union BF16Frag { v16bf v; v8bf h[2]; };

// ---------- CDNA5 feature detection (graceful fallback) ----------
#if __has_builtin(__builtin_amdgcn_tensor_load_to_lds) && \
    __has_builtin(__builtin_amdgcn_s_wait_tensorcnt)
#define USE_TDM 1
#endif

__device__ __forceinline__ v8f wmma_bf16(v16bf a, v16bf b, v8f c) {
  // D = A(16x32 bf16) x B(32x16 bf16) + C(16x16 f32)
  return __builtin_amdgcn_wmma_f32_16x16x32_bf16(
      false, a, false, b, (short)0, c, false, false);
}

#ifdef USE_TDM
// Issue a 2D TDM load: tile (tile_x elems) x (tile_y rows), 2-byte elements,
// row stride stride_elems, from gsrc into LDS byte offset lds_off.
__device__ __forceinline__ void tdm_load_2d(unsigned lds_off, const void* gsrc,
                                            int tile_x, int tile_y,
                                            int stride_elems) {
  unsigned long long ga = (unsigned long long)gsrc;
  uint4a g0;
  g0[0] = 1u;                                     // count=1 (valid D#)
  g0[1] = lds_off;                                // lds_addr (bytes)
  g0[2] = (unsigned)ga;                           // global_addr[31:0]
  g0[3] = (unsigned)((ga >> 32) & 0x01FFFFFFu) |  // global_addr[56:32]
          0x80000000u;                            // type=2 ("image")
  int8a g1;
  g1[0] = 0x10000;          // wg_mask=0, data_size=1 (2 bytes)
  g1[1] = tile_x << 16;     // tensor_dim0 (low 16)
  g1[2] = tile_y << 16;     // tensor_dim0 hi=0 | tensor_dim1 (low 16)
  g1[3] = tile_x << 16;     // tensor_dim1 hi=0 | tile_dim0
  g1[4] = tile_y;           // tile_dim1 | tile_dim2=0
  g1[5] = stride_elems;     // tensor_dim0_stride[31:0]
  g1[6] = 0;                // stride hi | tensor_dim1_stride lo
  g1[7] = 0;
  int4a z4 = {0, 0, 0, 0};
#if __clang_major__ >= 23
  int8a z8 = {0, 0, 0, 0, 0, 0, 0, 0};
  __builtin_amdgcn_tensor_load_to_lds(g0, g1, z4, z4, z8, 0);
#else
  __builtin_amdgcn_tensor_load_to_lds(g0, g1, z4, z4, 0);
#endif
}
#endif

// ============================================================================
// Generic WMMA GEMM:  C(MxN) = A(MxK) * B(KxN)  [+ bias]
// A: fp32 or bf16 (A_F32), B: fp32 (converted to bf16 on stage)
// C: bf16 (OUT_BF16) or fp32 + bias
// Block: 256 threads (8 waves). Tile: BM=128, BN=128, BK=32.
// ============================================================================
template <bool A_F32, bool OUT_BF16>
__global__ __launch_bounds__(256) void gemm_wmma(
    const void* __restrict__ Aptr, const float* __restrict__ B,
    void* __restrict__ Cptr, const float* __restrict__ bias,
    int M, int N, int K) {
  __shared__ bf16 As[128][32];      // [m][k]
  __shared__ bf16 Bs[128][32];      // transposed: [n][k]

  const float* Af = (const float*)Aptr;
  const bf16*  Ab = (const bf16*)Aptr;

  const int tid  = threadIdx.x;
  const int wid  = tid >> 5;
  const int lane = tid & 31;
  const int l16  = lane & 15;
  const int hh   = lane >> 4;

  const int row0 = blockIdx.y * 128;
  const int col0 = blockIdx.x * 128;
  const int wm   = (wid & 3) * 32;
  const int wn   = (wid >> 2) * 64;

  v8f acc[2][4] = {};

  for (int k0 = 0; k0 < K; k0 += 32) {
    // ---- stage A tile (128x32) ----
#pragma unroll
    for (int p = 0; p < 4; ++p) {
      int e  = (p * 256 + tid) * 4;
      int r  = e >> 5;
      int kk = e & 31;
      if constexpr (A_F32) {
        v4f32 f = *reinterpret_cast<const v4f32*>(
            &Af[(size_t)(row0 + r) * K + k0 + kk]);
        *reinterpret_cast<v4bf*>(&As[r][kk]) = __builtin_convertvector(f, v4bf);
      } else {
        *reinterpret_cast<v4bf*>(&As[r][kk]) =
            *reinterpret_cast<const v4bf*>(&Ab[(size_t)(row0 + r) * K + k0 + kk]);
      }
    }
    // ---- stage B tile transposed (Bs[n][k] = B[k0+k][col0+n]) ----
#pragma unroll
    for (int p = 0; p < 4; ++p) {
      int e  = (p * 256 + tid) * 4;
      int kk = e >> 7;
      int n  = e & 127;
      v4f32 f = *reinterpret_cast<const v4f32*>(
          &B[(size_t)(k0 + kk) * N + col0 + n]);
      v4bf b4 = __builtin_convertvector(f, v4bf);
      Bs[n + 0][kk] = b4[0]; Bs[n + 1][kk] = b4[1];
      Bs[n + 2][kk] = b4[2]; Bs[n + 3][kk] = b4[3];
    }
    if (k0 + 32 < K) {  // prefetch next tiles (global_prefetch_b8)
      if constexpr (A_F32)
        __builtin_prefetch(&Af[(size_t)(row0 + (tid >> 3)) * K + k0 + 32], 0, 3);
      else
        __builtin_prefetch(&Ab[(size_t)(row0 + (tid >> 3)) * K + k0 + 32], 0, 3);
      __builtin_prefetch(&B[(size_t)(k0 + 32 + (tid >> 5)) * N + col0], 0, 3);
    }
    __syncthreads();

    v16bf afr[2], bfr[4];
#pragma unroll
    for (int mt = 0; mt < 2; ++mt) {
      const bf16* rp = &As[wm + mt * 16 + l16][0];
      BF16Frag f;
      f.h[0] = *reinterpret_cast<const v8bf*>(rp + hh * 8);
      f.h[1] = *reinterpret_cast<const v8bf*>(rp + 16 + hh * 8);
      afr[mt] = f.v;
    }
#pragma unroll
    for (int nt = 0; nt < 4; ++nt) {
      const bf16* rp = &Bs[wn + nt * 16 + l16][0];
      BF16Frag f;
      f.h[0] = *reinterpret_cast<const v8bf*>(rp + hh * 8);
      f.h[1] = *reinterpret_cast<const v8bf*>(rp + hh * 8 + 16);
      bfr[nt] = f.v;
    }
#pragma unroll
    for (int mt = 0; mt < 2; ++mt)
#pragma unroll
      for (int nt = 0; nt < 4; ++nt)
        acc[mt][nt] = wmma_bf16(afr[mt], bfr[nt], acc[mt][nt]);
    __syncthreads();
  }

  // ---- epilogue: D layout = (M = r + 8*half, N = lane&15) ----
#pragma unroll
  for (int mt = 0; mt < 2; ++mt)
#pragma unroll
    for (int nt = 0; nt < 4; ++nt) {
      int gr0 = row0 + wm + mt * 16 + hh * 8;
      int gc  = col0 + wn + nt * 16 + l16;
#pragma unroll
      for (int r = 0; r < 8; ++r) {
        float v = acc[mt][nt][r];
        size_t off = (size_t)(gr0 + r) * N + gc;
        if constexpr (OUT_BF16)
          ((bf16*)Cptr)[off] = (bf16)v;
        else
          ((float*)Cptr)[off] = v + bias[gc];
      }
    }
}

// ============================================================================
// RoPE (in place, bf16)
// ============================================================================
__global__ __launch_bounds__(256) void rope_kernel(bf16* buf, int rowStride,
                                                   int ropeCols, long totalPairs) {
  long i = (long)blockIdx.x * 256 + threadIdx.x;
  if (i >= totalPairs) return;
  int  pairsPerRow = ropeCols >> 1;
  long row = i / pairsPerRow;
  int  c   = (int)(i % pairsPerRow) * 2;
  int  n   = (int)(row & 255);
  int  pa  = (c & 63) >> 1;
  float t  = (pa < 16) ? (float)(n >> 4) : (float)(n & 15);
  float th = t * __expf(-(float)(pa & 15) * 0.5756462732485114f); // ln(1e4)/16
  float cs, sn;
  __sincosf(th, &cs, &sn);
  bf16* p = buf + row * (size_t)rowStride + c;
  float x0 = (float)p[0], x1 = (float)p[1];
  p[0] = (bf16)(x0 * cs - x1 * sn);
  p[1] = (bf16)(x1 * cs + x0 * sn);
}

// ============================================================================
// Attention: one block per (b,h). 8 waves; wave w owns query rows [w*32,+32).
// K tile (256x64, row stride 2048) and raw V tile staged via TDM
// tensor_load_to_lds (TENSORcnt); V then LDS-transposed into Vt.
// Dynamic LDS: Ks 32KB @0 | Vt 32KB @32768 | P 64KB @65536 | Vs 32KB @131072.
// ============================================================================
__global__ __launch_bounds__(256) void attn_wmma(
    const bf16* __restrict__ qh,   // (B*256, 1024) bf16, RoPE applied
    const bf16* __restrict__ kvp,  // (B*256, 2048) bf16, K half RoPE applied
    bf16* __restrict__ out) {      // (B*256, 1024) bf16
  extern __shared__ char smem[];
  bf16 (*Ks)[64]  = (bf16(*)[64])smem;              // [n][d]
  bf16 (*Vt)[256] = (bf16(*)[256])(smem + 32768);   // [d][n]
  bf16* Pbase     = (bf16*)(smem + 65536);          // 8 x [16][256]

  const int b = blockIdx.x, h = blockIdx.y;
  const int tid = threadIdx.x, wid = tid >> 5, lane = tid & 31;
  const int l16 = lane & 15, hh = lane >> 4;
  const size_t kvrow0 = (size_t)b * 256 * 2048;
  const size_t qrow0  = (size_t)b * 256 * 1024;
  const int colk = h * 64, colv = 1024 + h * 64;

#ifdef USE_TDM
  // ---- K and raw-V tiles via Tensor Data Mover (wave 0 issues, waits) ----
  if (wid == 0) {
    tdm_load_2d(0u,      &kvp[kvrow0 + colk], 64, 256, 2048);  // -> Ks
    tdm_load_2d(131072u, &kvp[kvrow0 + colv], 64, 256, 2048);  // -> Vs
    __builtin_amdgcn_s_wait_tensorcnt(0);
  }
  __syncthreads();
  // ---- LDS -> LDS transpose Vs[n][d] -> Vt[d][n] (all threads) ----
  {
    bf16 (*Vs)[64] = (bf16(*)[64])(smem + 131072);
#pragma unroll
    for (int ps = 0; ps < 16; ++ps) {
      int e = (ps * 256 + tid) * 4;
      int n = e >> 6, d = e & 63;
      v4bf vv = *reinterpret_cast<const v4bf*>(&Vs[n][d]);
      Vt[d + 0][n] = vv[0]; Vt[d + 1][n] = vv[1];
      Vt[d + 2][n] = vv[2]; Vt[d + 3][n] = vv[3];
    }
  }
  __syncthreads();
#else
  // ---- fallback: manual global -> LDS staging ----
#pragma unroll
  for (int ps = 0; ps < 16; ++ps) {
    int e = (ps * 256 + tid) * 4;
    int n = e >> 6, d = e & 63;
    *reinterpret_cast<v4bf*>(&Ks[n][d]) = *reinterpret_cast<const v4bf*>(
        &kvp[kvrow0 + (size_t)n * 2048 + colk + d]);
    v4bf vv = *reinterpret_cast<const v4bf*>(
        &kvp[kvrow0 + (size_t)n * 2048 + colv + d]);
    Vt[d + 0][n] = vv[0]; Vt[d + 1][n] = vv[1];
    Vt[d + 2][n] = vv[2]; Vt[d + 3][n] = vv[3];
  }
  __syncthreads();
#endif

  bf16* P = Pbase + (size_t)wid * 16 * 256;  // per-wave scratch [16][256]
  const int m0 = wid * 32;

#pragma unroll
  for (int mt = 0; mt < 2; ++mt) {
    const int qm = m0 + mt * 16;

    v16bf qf[2];
#pragma unroll
    for (int kc = 0; kc < 2; ++kc) {
      const bf16* qp = &qh[qrow0 + (size_t)(qm + l16) * 1024 + colk + kc * 32];
      BF16Frag f;
      f.h[0] = *reinterpret_cast<const v8bf*>(qp + hh * 8);
      f.h[1] = *reinterpret_cast<const v8bf*>(qp + 16 + hh * 8);
      qf[kc] = f.v;
    }

    // ---- S = Q K^T : 32 WMMAs ----
    v8f s[16] = {};
#pragma unroll
    for (int kc = 0; kc < 2; ++kc)
#pragma unroll
      for (int j = 0; j < 16; ++j) {
        const bf16* kp = &Ks[j * 16 + l16][kc * 32 + hh * 8];
        BF16Frag f;
        f.h[0] = *reinterpret_cast<const v8bf*>(kp);
        f.h[1] = *reinterpret_cast<const v8bf*>(kp + 16);
        s[j] = wmma_bf16(qf[kc], f.v, s[j]);
      }

    // ---- exact softmax; row lives on one 16-lane half ----
#pragma unroll
    for (int r = 0; r < 8; ++r) {
      float mx = -3.0e38f;
#pragma unroll
      for (int j = 0; j < 16; ++j) mx = fmaxf(mx, s[j][r]);
      for (int o = 8; o >= 1; o >>= 1) mx = fmaxf(mx, __shfl_xor(mx, o, 16));
      float pv[16], sum = 0.f;
#pragma unroll
      for (int j = 0; j < 16; ++j) {
        pv[j] = __expf((s[j][r] - mx) * 0.125f);  // scale = 1/sqrt(64)
        sum += pv[j];
      }
      for (int o = 8; o >= 1; o >>= 1) sum += __shfl_xor(sum, o, 16);
      float inv = 1.0f / sum;
      int lr = r + 8 * hh;
#pragma unroll
      for (int j = 0; j < 16; ++j)
        P[(size_t)lr * 256 + j * 16 + l16] = (bf16)(pv[j] * inv);
    }
    // per-wave LDS region; same-wave DS ops are in order -> no barrier needed

    // ---- O = P V : 32 WMMAs ----
    v8f o[4] = {};
#pragma unroll
    for (int c = 0; c < 8; ++c) {
      const bf16* pp = &P[(size_t)l16 * 256 + c * 32];
      BF16Frag f;
      f.h[0] = *reinterpret_cast<const v8bf*>(pp + hh * 8);
      f.h[1] = *reinterpret_cast<const v8bf*>(pp + 16 + hh * 8);
      v16bf pf = f.v;
#pragma unroll
      for (int nt = 0; nt < 4; ++nt) {
        const bf16* vp = &Vt[nt * 16 + l16][c * 32 + hh * 8];
        BF16Frag g;
        g.h[0] = *reinterpret_cast<const v8bf*>(vp);
        g.h[1] = *reinterpret_cast<const v8bf*>(vp + 16);
        o[nt] = wmma_bf16(pf, g.v, o[nt]);
      }
    }

#pragma unroll
    for (int nt = 0; nt < 4; ++nt) {
      int gc = colk + nt * 16 + l16;
#pragma unroll
      for (int r = 0; r < 8; ++r)
        out[qrow0 + (size_t)(qm + r + 8 * hh) * 1024 + gc] = (bf16)o[nt][r];
    }
  }
}

// ============================================================================
// Launcher
// ============================================================================
extern "C" void kernel_launch(void* const* d_in, const int* in_sizes, int n_in,
                              void* d_out, int out_size, void* d_ws, size_t ws_size,
                              hipStream_t stream) {
  const float* q     = (const float*)d_in[0];  // (64,256,1024)
  const float* kv    = (const float*)d_in[1];  // (64,256,768)
  const float* Wq    = (const float*)d_in[2];  // (1024,1024)
  const float* Wkv   = (const float*)d_in[3];  // (768,2048)
  const float* Wproj = (const float*)d_in[4];  // (1024,1024)
  const float* bproj = (const float*)d_in[5];  // (1024,)
  float* out = (float*)d_out;

  const int M = 64 * 256;  // 16384
  char* ws = (char*)d_ws;
  bf16* qh   = (bf16*)ws;                                   // 16384x1024 bf16
  bf16* kvp  = (bf16*)(ws + (size_t)32 * 1024 * 1024);      // 16384x2048 bf16
  bf16* attn = (bf16*)(ws + (size_t)96 * 1024 * 1024);      // 16384x1024 bf16

  dim3 blk(256);

  gemm_wmma<true, true><<<dim3(1024 / 128, M / 128), blk, 0, stream>>>(
      q, Wq, qh, nullptr, M, 1024, 1024);
  gemm_wmma<true, true><<<dim3(2048 / 128, M / 128), blk, 0, stream>>>(
      kv, Wkv, kvp, nullptr, M, 2048, 768);

  long pairs = (long)M * 512;
  rope_kernel<<<(pairs + 255) / 256, blk, 0, stream>>>(qh, 1024, 1024, pairs);
  rope_kernel<<<(pairs + 255) / 256, blk, 0, stream>>>(kvp, 2048, 1024, pairs);

  size_t lds = 160 * 1024;  // Ks + Vt + P scratch + Vs staging
  (void)hipFuncSetAttribute((const void*)attn_wmma,
                            hipFuncAttributeMaxDynamicSharedMemorySize,
                            (int)lds);
  attn_wmma<<<dim3(64, 16), blk, lds, stream>>>(qh, kvp, attn);

  gemm_wmma<false, false><<<dim3(1024 / 128, M / 128), blk, 0, stream>>>(
      attn, Wproj, out, bproj, M, 1024, 1024);
}
